// GPTModel_77412490543222
// MI455X (gfx1250) — matrix operand order repaired and checked by reading, not compiled, not run on hardware
//
#include <hip/hip_runtime.h>
#include <hip/hip_bf16.h>
#include <math.h>

// ---------------------------------------------------------------------------
// GPT-2-ish tiny model forward on MI455X (gfx1250).
//   V=50257 B=16 T=256 D=64 H=4 HS=16 L=4 F=256,  M = B*T = 4096 rows.
// Head GEMM + log-softmax dominate (823 MB logits write + re-read);
// all GEMMs run through v_wmma_f32_16x16x32_bf16 with fp32 accumulation.
// Weights are pre-converted fp32 -> bf16 *transposed* to [N,K] so each lane's
// B operand is one contiguous 32-byte v16bf load (B order used here: K = e+16*half,
// N = lane%16).  A operand ([M,K] row-major bf16) is two 16-byte loads per
// lane (same K order as B: K = e+16*half).
// ---------------------------------------------------------------------------

#define VOCAB 50257
#define BB 16
#define TT 256
#define DD 64
#define HH 4
#define HSS 16
#define LL 4
#define FF 256
#define MM (BB * TT)          // 4096 rows
#define NQKV 192              // q|k|v concatenated per row
#define VPAD 50272            // 50257 rounded up to multiple of 16

typedef __bf16 bf16_t;
typedef __attribute__((ext_vector_type(16))) __bf16 v16bf;
typedef __attribute__((ext_vector_type(8)))  __bf16 v8bf;
typedef __attribute__((ext_vector_type(8)))  float  v8f;

// ---------------------------------------------------------------------------
// fp32 [K,N] -> bf16 [Nout,K] transpose (zero-pad rows n >= N).
// ---------------------------------------------------------------------------
__global__ void __launch_bounds__(256)
transpose_bf16_kernel(const float* __restrict__ W, bf16_t* __restrict__ out,
                      int K, int N, int Nout) {
  int tid = blockIdx.x * 256 + threadIdx.x;
  int total = Nout * K;
  if (tid >= total) return;
  int n = tid / K;
  int k = tid - n * K;
  float v = (n < N) ? W[(size_t)k * N + n] : 0.0f;
  out[(size_t)n * K + k] = (bf16_t)v;
}

// ---------------------------------------------------------------------------
// x[b,t,:] = tok_emb[idx[b,t]] + pos_emb[t]
// ---------------------------------------------------------------------------
__global__ void __launch_bounds__(256)
embed_kernel(const int* __restrict__ idx, const float* __restrict__ tok,
             const float* __restrict__ pos, float* __restrict__ x) {
  int tid = blockIdx.x * 256 + threadIdx.x;
  if (tid >= MM * DD) return;
  int row = tid / DD;
  int d   = tid - row * DD;
  int t   = row % TT;
  x[tid] = tok[(size_t)idx[row] * DD + d] + pos[t * DD + d];
}

// ---------------------------------------------------------------------------
// LayerNorm over D=64, one wave (32 lanes * 2 elems) per row, bf16 output.
// ---------------------------------------------------------------------------
__global__ void __launch_bounds__(256)
ln_kernel(const float* __restrict__ x, const float* __restrict__ g,
          const float* __restrict__ b, bf16_t* __restrict__ out, int rows) {
  int lane = threadIdx.x & 31;
  int wv   = threadIdx.x >> 5;
  int row  = blockIdx.x * 8 + wv;
  if (row >= rows) return;
  const float* xr = x + (size_t)row * DD;
  float a0 = xr[2 * lane], a1 = xr[2 * lane + 1];
  float s  = a0 + a1;
  float sq = a0 * a0 + a1 * a1;
  for (int m = 16; m >= 1; m >>= 1) {
    s  += __shfl_xor(s,  m, 32);
    sq += __shfl_xor(sq, m, 32);
  }
  float mu  = s * (1.0f / DD);
  float var = sq * (1.0f / DD) - mu * mu;
  float inv = rsqrtf(var + 1e-5f);
  out[(size_t)row * DD + 2 * lane]     = (bf16_t)((a0 - mu) * inv * g[2 * lane]     + b[2 * lane]);
  out[(size_t)row * DD + 2 * lane + 1] = (bf16_t)((a1 - mu) * inv * g[2 * lane + 1] + b[2 * lane + 1]);
}

// ---------------------------------------------------------------------------
// WMMA bf16 GEMM: D = A[M,K](bf16) * Bt[N,K]^T(bf16) (+bias)(+residual)(relu?)
// One 16x16 tile per wave, 8 waves per block.  fp32 and/or bf16 outputs.
// ---------------------------------------------------------------------------
__global__ void __launch_bounds__(256)
wmma_gemm_bf16(const bf16_t* __restrict__ A, const bf16_t* __restrict__ Bt,
               const float* __restrict__ bias, const float* __restrict__ Cres,
               float* __restrict__ Dout, bf16_t* __restrict__ Dbf,
               int Mt, int Nt, int Nreal, int K, int relu) {
  int lane = threadIdx.x & 31;
  int wv   = threadIdx.x >> 5;
  int tile = blockIdx.x * 8 + wv;
  if (tile >= Mt * Nt) return;           // wave-uniform: EXEC all-ones at WMMA
  int m0   = (tile / Nt) << 4;
  int n0   = (tile % Nt) << 4;
  int half = lane >> 4;                  // 0..1
  int ml   = lane & 15;                  // M (for A) / N (for B,C,D)

  const bf16_t* pa = A  + (size_t)(m0 + ml) * K;             // row-major [M,K]
  const bf16_t* pb = Bt + (size_t)(n0 + ml) * K + 16 * half; // transposed [N,K]

  v8f acc = {};
  for (int k0 = 0; k0 < K; k0 += 32) {
    union { v16bf v; v8bf h[2]; } a;
    a.h[0] = *(const v8bf*)(pa + k0 + 16 * half);       // K = 16*half  .. +7
    a.h[1] = *(const v8bf*)(pa + k0 + 16 * half + 8);   // K = 16*half+8.. +15
    v16bf bvec = *(const v16bf*)(pb + k0);              // K = 16*half  .. +15
    acc = __builtin_amdgcn_wmma_f32_16x16x32_bf16(
        false, a.v, false, bvec, (short)0, acc, false, false);
  }

  int n = n0 + ml;
  if (n < Nreal) {
    float bval = bias ? bias[n] : 0.0f;
    for (int r = 0; r < 8; ++r) {
      int m = m0 + r + 8 * half;                        // C/D layout
      size_t off = (size_t)m * Nreal + n;
      float v = acc[r] + bval;
      if (Cres) v += Cres[off];
      if (relu) v = fmaxf(v, 0.0f);
      if (Dout) Dout[off] = v;
      if (Dbf)  Dbf[off]  = (bf16_t)v;
    }
  }
}

// ---------------------------------------------------------------------------
// Causal attention per (b,h): scores = q k^T (no 1/sqrt(hs) scaling, matching
// the reference), two-pass softmax, o = attn @ v.  K/V staged in LDS.
// Output bf16 at obf[(b*T+t)*64 + h*16 + s] (heads concatenated).
// ---------------------------------------------------------------------------
__global__ void __launch_bounds__(TT)
attn_kernel(const float* __restrict__ qkv, bf16_t* __restrict__ obf) {
  __shared__ float ks[TT][HSS];
  __shared__ float vs[TT][HSS];
  int bh = blockIdx.x;
  int b  = bh / HH;
  int h  = bh % HH;
  int t  = threadIdx.x;
  const float* base = qkv + (size_t)(b * TT + t) * NQKV;
  float q[HSS];
  for (int s = 0; s < HSS; ++s) {
    q[s]     = base[h * HSS + s];
    ks[t][s] = base[64  + h * HSS + s];
    vs[t][s] = base[128 + h * HSS + s];
  }
  __syncthreads();
  float mx = -3.4e38f;
  for (int u = 0; u <= t; ++u) {
    float sc = 0.0f;
    for (int s = 0; s < HSS; ++s) sc += q[s] * ks[u][s];
    mx = fmaxf(mx, sc);
  }
  float sum = 0.0f;
  float o[HSS];
  for (int s = 0; s < HSS; ++s) o[s] = 0.0f;
  for (int u = 0; u <= t; ++u) {
    float sc = 0.0f;
    for (int s = 0; s < HSS; ++s) sc += q[s] * ks[u][s];
    float e = __expf(sc - mx);
    sum += e;
    for (int s = 0; s < HSS; ++s) o[s] += e * vs[u][s];
  }
  float inv = 1.0f / sum;
  bf16_t* dst = obf + (size_t)(b * TT + t) * DD + h * HSS;
  for (int s = 0; s < HSS; ++s) dst[s] = (bf16_t)(o[s] * inv);
}

// ---------------------------------------------------------------------------
// Per-row -log p[target] over V=50257 (streaming max, then sum-exp).
// ---------------------------------------------------------------------------
__global__ void __launch_bounds__(256)
rowloss_kernel(const float* __restrict__ logits, const int* __restrict__ tgt,
               float* __restrict__ row_loss) {
  __shared__ float red[256];
  int r = blockIdx.x;
  int tid = threadIdx.x;
  const float* row = logits + (size_t)r * VOCAB;
  float mx = -3.4e38f;
  for (int i = tid; i < VOCAB; i += 256) mx = fmaxf(mx, row[i]);
  red[tid] = mx;
  __syncthreads();
  for (int m = 128; m >= 1; m >>= 1) {
    if (tid < m) red[tid] = fmaxf(red[tid], red[tid + m]);
    __syncthreads();
  }
  mx = red[0];
  __syncthreads();
  float s = 0.0f;
  for (int i = tid; i < VOCAB; i += 256) s += __expf(row[i] - mx);
  red[tid] = s;
  __syncthreads();
  for (int m = 128; m >= 1; m >>= 1) {
    if (tid < m) red[tid] += red[tid + m];
    __syncthreads();
  }
  if (tid == 0)
    row_loss[r] = (mx + __logf(red[0])) - row[tgt[r]];
}

__global__ void __launch_bounds__(256)
finalloss_kernel(const float* __restrict__ row_loss, float* __restrict__ loss_out) {
  __shared__ float red[256];
  float s = 0.0f;
  for (int i = threadIdx.x; i < MM; i += 256) s += row_loss[i];
  red[threadIdx.x] = s;
  __syncthreads();
  for (int m = 128; m >= 1; m >>= 1) {
    if (threadIdx.x < m) red[threadIdx.x] += red[threadIdx.x + m];
    __syncthreads();
  }
  if (threadIdx.x == 0) *loss_out = red[0] * (1.0f / MM);
}

// ---------------------------------------------------------------------------
extern "C" void kernel_launch(void* const* d_in, const int* in_sizes, int n_in,
                              void* d_out, int out_size, void* d_ws, size_t ws_size,
                              hipStream_t stream) {
  (void)in_sizes; (void)n_in; (void)out_size; (void)ws_size;
  const int*   idx     = (const int*)d_in[0];
  const int*   targets = (const int*)d_in[1];
  const float* tok_emb = (const float*)d_in[2];
  const float* pos_emb = (const float*)d_in[3];
  const float* Wq      = (const float*)d_in[4];
  const float* Wk      = (const float*)d_in[5];
  const float* Wv      = (const float*)d_in[6];
  const float* Wo      = (const float*)d_in[7];
  const float* bo      = (const float*)d_in[8];
  const float* ln1_g   = (const float*)d_in[9];
  const float* ln1_b   = (const float*)d_in[10];
  const float* ln2_g   = (const float*)d_in[11];
  const float* ln2_b   = (const float*)d_in[12];
  const float* W1      = (const float*)d_in[13];
  const float* b1      = (const float*)d_in[14];
  const float* W2      = (const float*)d_in[15];
  const float* b2      = (const float*)d_in[16];
  const float* lnf_g   = (const float*)d_in[17];
  const float* lnf_b   = (const float*)d_in[18];
  const float* Whead   = (const float*)d_in[19];
  const float* bhead   = (const float*)d_in[20];
  float* out = (float*)d_out;

  // ---- workspace carve-up (256B aligned) ----
  char* wsp = (char*)d_ws;
  auto alloc = [&](size_t bytes) {
    char* p = wsp;
    wsp += (bytes + 255) & ~(size_t)255;
    return p;
  };
  float*  x        = (float*) alloc((size_t)MM * DD * 4);
  bf16_t* hbf      = (bf16_t*)alloc((size_t)MM * DD * 2);
  float*  qkv      = (float*) alloc((size_t)MM * NQKV * 4);
  bf16_t* obf      = (bf16_t*)alloc((size_t)MM * DD * 2);
  bf16_t* ubf      = (bf16_t*)alloc((size_t)MM * FF * 2);
  bf16_t* xfbf     = (bf16_t*)alloc((size_t)MM * DD * 2);
  bf16_t* Bqkv     = (bf16_t*)alloc((size_t)LL * NQKV * DD * 2);
  bf16_t* Bo       = (bf16_t*)alloc((size_t)LL * DD * DD * 2);
  bf16_t* B1t      = (bf16_t*)alloc((size_t)LL * FF * DD * 2);
  bf16_t* B2t      = (bf16_t*)alloc((size_t)LL * DD * FF * 2);
  bf16_t* Bhead    = (bf16_t*)alloc((size_t)VPAD * DD * 2);
  float*  row_loss = (float*) alloc((size_t)MM * 4);

  auto tr = [&](const float* W, bf16_t* o, int K, int N, int Nout) {
    int total = Nout * K;
    transpose_bf16_kernel<<<(total + 255) / 256, 256, 0, stream>>>(W, o, K, N, Nout);
  };
  auto gemm = [&](const bf16_t* A, const bf16_t* Bt, const float* bias,
                  const float* Cres, float* Dout, bf16_t* Dbf,
                  int Mrows, int N16, int Nreal, int K, int relu) {
    int Mt = Mrows / 16, Nt = N16 / 16;
    int tiles = Mt * Nt;
    wmma_gemm_bf16<<<(tiles + 7) / 8, 256, 0, stream>>>(
        A, Bt, bias, Cres, Dout, Dbf, Mt, Nt, Nreal, K, relu);
  };

  // ---- weight conversion: fp32 [K,N] -> bf16 [N,K] (transposed) ----
  for (int l = 0; l < LL; ++l) {
    for (int h = 0; h < HH; ++h) {
      tr(Wq + (size_t)(l * HH + h) * DD * HSS, Bqkv + (size_t)l * NQKV * DD + (0   + h * HSS) * DD, DD, HSS, HSS);
      tr(Wk + (size_t)(l * HH + h) * DD * HSS, Bqkv + (size_t)l * NQKV * DD + (64  + h * HSS) * DD, DD, HSS, HSS);
      tr(Wv + (size_t)(l * HH + h) * DD * HSS, Bqkv + (size_t)l * NQKV * DD + (128 + h * HSS) * DD, DD, HSS, HSS);
    }
    tr(Wo + (size_t)l * DD * DD, Bo  + (size_t)l * DD * DD, DD, DD, DD);
    tr(W1 + (size_t)l * DD * FF, B1t + (size_t)l * FF * DD, DD, FF, FF);
    tr(W2 + (size_t)l * FF * DD, B2t + (size_t)l * DD * FF, FF, DD, DD);
  }
  tr(Whead, Bhead, DD, VOCAB, VPAD);

  // ---- forward ----
  embed_kernel<<<(MM * DD + 255) / 256, 256, 0, stream>>>(idx, tok_emb, pos_emb, x);

  for (int l = 0; l < LL; ++l) {
    ln_kernel<<<MM / 8, 256, 0, stream>>>(x, ln1_g + l * DD, ln1_b + l * DD, hbf, MM);
    gemm(hbf, Bqkv + (size_t)l * NQKV * DD, nullptr, nullptr, qkv, nullptr,
         MM, NQKV, NQKV, DD, 0);
    attn_kernel<<<BB * HH, TT, 0, stream>>>(qkv, obf);
    gemm(obf, Bo + (size_t)l * DD * DD, bo + l * DD, x, x, nullptr,
         MM, DD, DD, DD, 0);
    ln_kernel<<<MM / 8, 256, 0, stream>>>(x, ln2_g + l * DD, ln2_b + l * DD, hbf, MM);
    gemm(hbf, B1t + (size_t)l * FF * DD, b1 + l * FF, nullptr, nullptr, ubf,
         MM, FF, FF, DD, 1);
    gemm(ubf, B2t + (size_t)l * DD * FF, b2 + l * DD, x, x, nullptr,
         MM, DD, DD, FF, 0);
  }

  ln_kernel<<<MM / 8, 256, 0, stream>>>(x, lnf_g, lnf_b, xfbf, MM);

  // Head GEMM: logits straight into d_out (store-bound: ~823 MB fp32).
  gemm(xfbf, Bhead, bhead, nullptr, out, nullptr, MM, VPAD, VOCAB, DD, 0);

  // Loss.
  rowloss_kernel<<<MM, 256, 0, stream>>>(out, targets, row_loss);
  finalloss_kernel<<<1, 256, 0, stream>>>(row_loss, out + (size_t)MM * VOCAB);
}
